// DPGraphConvolution_14156212208279
// MI455X (gfx1250) — matrix operand order repaired and checked by reading, not compiled, not run on hardware
//
#include <hip/hip_runtime.h>

// ---------------------------------------------------------------------------
// out[b,i,:] = sum_j adj[b,i,j] * (text[b,j,:] @ W) + bias       (B=8,L=4096,F=64)
//
// Strategy (MI455X / gfx1250):
//   fp32 emulated via split-bf16 (hi+lo limbs) on v_wmma_f32_16x16x32_bf16.
//   3 WMMAs per tile (hi*hi + hi*lo + lo*hi) -> ~1e-5 rel error, while the
//   kernel stays HBM-bound on the 537MB adj read (~24us floor @ 23.3TB/s).
// ---------------------------------------------------------------------------

#define Bsz 8
#define Lg  4096
#define Fg  64

typedef __attribute__((ext_vector_type(16))) __bf16 v16bf;
typedef __attribute__((ext_vector_type(8)))  float  v8f;

union Frag16 {
    unsigned int   u32[8];
    unsigned short u16[16];
    v16bf          v;
};

// Split a pair of fp32 into packed bf16 hi-limbs and lo-limbs.
// hi = truncate-to-bf16(x) (so lo = x - hi is exact by Sterbenz),
// lo = truncate-to-bf16(x - hi).  Packing via v_perm_b32.
__device__ __forceinline__ void split2(float x0, float x1,
                                       unsigned int& hi, unsigned int& lo) {
    unsigned int u0 = __float_as_uint(x0);
    unsigned int u1 = __float_as_uint(x1);
    hi = __builtin_amdgcn_perm(u1, u0, 0x07060302u); // {u1[31:16], u0[31:16]}
    float l0 = x0 - __uint_as_float(u0 & 0xFFFF0000u);
    float l1 = x1 - __uint_as_float(u1 & 0xFFFF0000u);
    lo = __builtin_amdgcn_perm(__float_as_uint(l1), __float_as_uint(l0), 0x07060302u);
}

// -------------------- kernel 1: split W -> WT_hi/lo[n][k] (bf16) ------------
__global__ __launch_bounds__(256)
void gconv_prepw(const float* __restrict__ W,
                 unsigned short* __restrict__ WT_hi,
                 unsigned short* __restrict__ WT_lo) {
    int t = blockIdx.x * blockDim.x + threadIdx.x;   // 4096 elements
    if (t < Fg * Fg) {
        int k = t >> 6;        // input feature (row of W)
        int n = t & 63;        // output feature (col of W)
        float x = W[t];
        unsigned int u = __float_as_uint(x);
        unsigned short hi = (unsigned short)(u >> 16);
        float lof = x - __uint_as_float(u & 0xFFFF0000u);
        unsigned short lo = (unsigned short)(__float_as_uint(lof) >> 16);
        WT_hi[n * Fg + k] = hi;
        WT_lo[n * Fg + k] = lo;
    }
}

// ---- kernel 2: hidden = text @ W (split-bf16 WMMA), stored transposed ------
// hiT_hi/lo[b][f][l]  so the main GEMM's B fragments are contiguous loads.
__global__ __launch_bounds__(256)
void gconv_hidden(const float* __restrict__ text,
                  const unsigned short* __restrict__ WT_hi,
                  const unsigned short* __restrict__ WT_lo,
                  unsigned short* __restrict__ hiT_hi,
                  unsigned short* __restrict__ hiT_lo) {
    const int lane = threadIdx.x & 31;
    const int w    = threadIdx.x >> 5;
    const int mt   = w >> 2;            // 0..1  (M sub-tile)
    const int nt   = w & 3;             // 0..3  (N sub-tile)
    const int b    = blockIdx.x >> 7;   // batch
    const int m0   = (blockIdx.x & 127) << 5;   // 32 rows per block
    const int lrow = lane & 15;
    const int hA   = (lane >> 4) << 3;  // A k-offset: 0 or 8
    const int hB   = (lane >> 4) << 3;  // B k-offset: 0 or 8

    const float* arow = text + ((size_t)(b * Lg + m0 + mt * 16 + lrow)) * Fg + hA;
    const unsigned short* bh = WT_hi + (size_t)(nt * 16 + lrow) * Fg + hB;
    const unsigned short* bl = WT_lo + (size_t)(nt * 16 + lrow) * Fg + hB;

    v8f acc = {0.f, 0.f, 0.f, 0.f, 0.f, 0.f, 0.f, 0.f};

#pragma unroll
    for (int k0 = 0; k0 < Fg; k0 += 32) {
        float4 a0 = *(const float4*)(arow + k0);
        float4 a1 = *(const float4*)(arow + k0 + 4);
        float4 a2 = *(const float4*)(arow + k0 + 16);
        float4 a3 = *(const float4*)(arow + k0 + 20);
        Frag16 ah, al;
        split2(a0.x, a0.y, ah.u32[0], al.u32[0]);
        split2(a0.z, a0.w, ah.u32[1], al.u32[1]);
        split2(a1.x, a1.y, ah.u32[2], al.u32[2]);
        split2(a1.z, a1.w, ah.u32[3], al.u32[3]);
        split2(a2.x, a2.y, ah.u32[4], al.u32[4]);
        split2(a2.z, a2.w, ah.u32[5], al.u32[5]);
        split2(a3.x, a3.y, ah.u32[6], al.u32[6]);
        split2(a3.z, a3.w, ah.u32[7], al.u32[7]);
        Frag16 bhh, bll;
        *(uint4*)&bhh.u32[0] = *(const uint4*)(bh + k0);
        *(uint4*)&bhh.u32[4] = *(const uint4*)(bh + k0 + 16);
        *(uint4*)&bll.u32[0] = *(const uint4*)(bl + k0);
        *(uint4*)&bll.u32[4] = *(const uint4*)(bl + k0 + 16);
        acc = __builtin_amdgcn_wmma_f32_16x16x32_bf16(false, ah.v, false, bhh.v,
                                                      (short)0, acc, false, false);
        acc = __builtin_amdgcn_wmma_f32_16x16x32_bf16(false, ah.v, false, bll.v,
                                                      (short)0, acc, false, false);
        acc = __builtin_amdgcn_wmma_f32_16x16x32_bf16(false, al.v, false, bhh.v,
                                                      (short)0, acc, false, false);
    }

    // D fragment: lane holds N = lane&15, M = m0 + mt*16 + 8*(lane>>4) + v.
    // Transposed store: 8 consecutive M values -> one 16B store per limb.
    const int n     = nt * 16 + lrow;
    const int mbase = m0 + mt * 16 + ((lane >> 4) << 3);
    union { unsigned int u32[4]; uint4 q; } oh, ol;
    split2(acc[0], acc[1], oh.u32[0], ol.u32[0]);
    split2(acc[2], acc[3], oh.u32[1], ol.u32[1]);
    split2(acc[4], acc[5], oh.u32[2], ol.u32[2]);
    split2(acc[6], acc[7], oh.u32[3], ol.u32[3]);
    size_t off = ((size_t)(b * Fg + n)) * Lg + mbase;
    *(uint4*)(hiT_hi + off) = oh.q;
    *(uint4*)(hiT_lo + off) = ol.q;
}

// ---- kernel 3: out = adj @ hidden + bias (split-bf16 WMMA, K = 4096) -------
__global__ __launch_bounds__(256)
void gconv_main(const float* __restrict__ adj,
                const unsigned short* __restrict__ hiT_hi,
                const unsigned short* __restrict__ hiT_lo,
                const float* __restrict__ bias,
                float* __restrict__ out) {
    const int lane = threadIdx.x & 31;
    const int w    = threadIdx.x >> 5;
    const int mt   = w >> 2;
    const int nt   = w & 3;
    const int b    = blockIdx.x >> 7;
    const int m0   = (blockIdx.x & 127) << 5;
    const int lrow = lane & 15;
    const int hA   = (lane >> 4) << 3;   // 0 or 8
    const int hB   = (lane >> 4) << 3;   // 0 or 8

    const float* arow = adj + ((size_t)(b * Lg + m0 + mt * 16 + lrow)) * Lg + hA;
    const unsigned short* bh = hiT_hi + ((size_t)(b * Fg + nt * 16 + lrow)) * Lg + hB;
    const unsigned short* bl = hiT_lo + ((size_t)(b * Fg + nt * 16 + lrow)) * Lg + hB;

    v8f acc = {0.f, 0.f, 0.f, 0.f, 0.f, 0.f, 0.f, 0.f};

    for (int k0 = 0; k0 < Lg; k0 += 32) {
        float4 a0 = *(const float4*)(arow + k0);
        float4 a1 = *(const float4*)(arow + k0 + 4);
        float4 a2 = *(const float4*)(arow + k0 + 16);
        float4 a3 = *(const float4*)(arow + k0 + 20);
        Frag16 ah, al;
        split2(a0.x, a0.y, ah.u32[0], al.u32[0]);
        split2(a0.z, a0.w, ah.u32[1], al.u32[1]);
        split2(a1.x, a1.y, ah.u32[2], al.u32[2]);
        split2(a1.z, a1.w, ah.u32[3], al.u32[3]);
        split2(a2.x, a2.y, ah.u32[4], al.u32[4]);
        split2(a2.z, a2.w, ah.u32[5], al.u32[5]);
        split2(a3.x, a3.y, ah.u32[6], al.u32[6]);
        split2(a3.z, a3.w, ah.u32[7], al.u32[7]);
        Frag16 bhh, bll;
        *(uint4*)&bhh.u32[0] = *(const uint4*)(bh + k0);
        *(uint4*)&bhh.u32[4] = *(const uint4*)(bh + k0 + 16);
        *(uint4*)&bll.u32[0] = *(const uint4*)(bl + k0);
        *(uint4*)&bll.u32[4] = *(const uint4*)(bl + k0 + 16);
        acc = __builtin_amdgcn_wmma_f32_16x16x32_bf16(false, ah.v, false, bhh.v,
                                                      (short)0, acc, false, false);
        acc = __builtin_amdgcn_wmma_f32_16x16x32_bf16(false, ah.v, false, bll.v,
                                                      (short)0, acc, false, false);
        acc = __builtin_amdgcn_wmma_f32_16x16x32_bf16(false, al.v, false, bhh.v,
                                                      (short)0, acc, false, false);
    }

    const float bn   = bias[nt * 16 + lrow];
    const int   mrow = m0 + mt * 16 + ((lane >> 4) << 3);
    float* orow = out + ((size_t)(b * Lg) + mrow) * Fg + nt * 16 + lrow;
#pragma unroll
    for (int v = 0; v < 8; ++v)
        orow[(size_t)v * Fg] = acc[v] + bn;
}

// ---------------------------------------------------------------------------
extern "C" void kernel_launch(void* const* d_in, const int* in_sizes, int n_in,
                              void* d_out, int out_size, void* d_ws, size_t ws_size,
                              hipStream_t stream) {
    const float* text = (const float*)d_in[0];   // [8,4096,64]
    const float* adj  = (const float*)d_in[1];   // [8,4096,4096]
    const float* W    = (const float*)d_in[2];   // [64,64]
    const float* bias = (const float*)d_in[3];   // [64]
    float* out = (float*)d_out;                  // [8,4096,64]

    // workspace: hiT_hi (4MB) | hiT_lo (4MB) | WT_hi (8KB) | WT_lo (8KB)
    const size_t hidN = (size_t)Bsz * Fg * Lg;
    unsigned short* hiT_hi = (unsigned short*)d_ws;
    unsigned short* hiT_lo = hiT_hi + hidN;
    unsigned short* WT_hi  = hiT_lo + hidN;
    unsigned short* WT_lo  = WT_hi + Fg * Fg;

    gconv_prepw <<<16,   256, 0, stream>>>(W, WT_hi, WT_lo);
    gconv_hidden<<<Bsz * (Lg / 32), 256, 0, stream>>>(text, WT_hi, WT_lo, hiT_hi, hiT_lo);
    gconv_main  <<<Bsz * (Lg / 32), 256, 0, stream>>>(adj, hiT_hi, hiT_lo, bias, out);
}
